// SelfAttnHead_42777874268560
// MI455X (gfx1250) — matrix operand order repaired and checked by reading, compiled, not run on hardware
//
#include <hip/hip_runtime.h>
#include <hip/hip_bf16.h>

// ---------------------------------------------------------------------------
// Self-attention head: out = softmax(mask + (xWq+bq)(xWk+bk)^T / sqrt(d)) (xWv+bv)
// B=4, L=2048, d=1024.  All GEMMs via v_wmma_f32_16x16x32_bf16 (wave32).
//
// GEMM core: block tile 128(M) x 64(N) x 32(K), 256 threads = 8 waves.
// Wave w owns rows [w*16, w*16+16) and all 64 cols: 1 A-fragment reused
// across 4 B-fragments -> 4 WMMAs per wave-iteration.
// LDS: A row-major [128][40], B transposed [n][k] = [64][40] so that BOTH
// fragment types are contiguous 16-byte ds_load_b128 reads.
// ---------------------------------------------------------------------------

typedef __attribute__((ext_vector_type(16))) __bf16 v16bf;
typedef __attribute__((ext_vector_type(8)))  float  v8f;
typedef __attribute__((ext_vector_type(8)))  __bf16 bf16x8;

#define EMBED 1024
#define SEQ   2048
#define BATCH 4
#define MROWS (BATCH * SEQ)   // 8192 flattened (b,l) rows

#define LDS_STRIDE 40         // 80 bytes: multiple of 16 -> aligned b128 rows

// ---- WMMA fragment loaders (per CDNA5 ISA 7.12.2 VGPR layouts) -------------
// A 16x32 (MxK): lane<16 -> row M=lane, K 0..7 & 16..23; lane>=16 -> K 8..15 & 24..31
__device__ __forceinline__ v16bf fragA(const __bf16* Abase, int lane) {
  const int m  = lane & 15;
  const int kb = (lane & 16) ? 8 : 0;
  const __bf16* r = Abase + m * LDS_STRIDE + kb;
  const bf16x8 lo = *reinterpret_cast<const bf16x8*>(r);       // K +0..7
  const bf16x8 hi = *reinterpret_cast<const bf16x8*>(r + 16);  // K +16..23
  return __builtin_shufflevector(lo, hi, 0, 1, 2, 3, 4, 5, 6, 7,
                                 8, 9, 10, 11, 12, 13, 14, 15);
}

// B 32x16 (KxN) stored transposed BsT[n][k]: lane<16 -> col N=lane, K 0..7 & 16..23;
// lane>=16 -> col N=lane-16, K 8..15 & 24..31.
__device__ __forceinline__ v16bf fragB(const __bf16* BsT, int nsub, int lane) {
  const int n  = (lane & 15) + nsub * 16;
  const int kb = (lane & 16) ? 8 : 0;
  const __bf16* r = BsT + n * LDS_STRIDE + kb;
  const bf16x8 lo = *reinterpret_cast<const bf16x8*>(r);
  const bf16x8 hi = *reinterpret_cast<const bf16x8*>(r + 16);
  return __builtin_shufflevector(lo, hi, 0, 1, 2, 3, 4, 5, 6, 7,
                                 8, 9, 10, 11, 12, 13, 14, 15);
}

__device__ __forceinline__ v8f wmma_bf16(v16bf a, v16bf b, v8f c) {
  return __builtin_amdgcn_wmma_f32_16x16x32_bf16(false, a, false, b, (short)0, c,
                                                 false, false);
}

__device__ __forceinline__ bf16x8 cvt8(const float4 a, const float4 b) {
  bf16x8 r;
  r[0] = (__bf16)a.x; r[1] = (__bf16)a.y; r[2] = (__bf16)a.z; r[3] = (__bf16)a.w;
  r[4] = (__bf16)b.x; r[5] = (__bf16)b.y; r[6] = (__bf16)b.z; r[7] = (__bf16)b.w;
  return r;
}

// ---------------------------------------------------------------------------
// Kernel 1: QKV projection.  Y[z] = (X @ W[z]^T + bias[z]) * scale[z] -> bf16
// grid = (EMBED/64, MROWS/128, 3); block = 256
// ---------------------------------------------------------------------------
__global__ __launch_bounds__(256) void qkv_proj_kernel(
    const float* __restrict__ X,
    const float* __restrict__ Wq, const float* __restrict__ bq,
    const float* __restrict__ Wk, const float* __restrict__ bk,
    const float* __restrict__ Wv, const float* __restrict__ bv,
    __bf16* __restrict__ out) {
  const int z = blockIdx.z;
  const float* W    = (z == 0) ? Wq : (z == 1) ? Wk : Wv;
  const float* bias = (z == 0) ? bq : (z == 1) ? bk : bv;
  const float scale = (z == 0) ? 0.03125f : 1.0f;   // 1/sqrt(1024) folded into Q
  __bf16* Y = out + (size_t)z * ((size_t)MROWS * EMBED);

  __shared__ __bf16 As[128][LDS_STRIDE];
  __shared__ __bf16 BsT[64][LDS_STRIDE];

  const int tid  = threadIdx.x;
  const int lane = tid & 31;
  const int w    = tid >> 5;            // wave 0..7 -> row group
  const int M0 = blockIdx.y * 128;
  const int N0 = blockIdx.x * 64;

  const int sa_m = tid >> 1;            // 0..127
  const int sa_k = (tid & 1) * 16;      // 0,16
  const int sb_n = tid >> 2;            // 0..63
  const int sb_k = (tid & 3) * 8;       // 0,8,16,24

  v8f acc[4] = {};

  const float* srcA = X + (size_t)(M0 + sa_m) * EMBED + sa_k;
  const float* srcB = W + (size_t)(N0 + sb_n) * EMBED + sb_k;

  for (int k0 = 0; k0 < EMBED; k0 += 32) {
    // Stage A tile (fp32 -> bf16), 16 elems/thread, vector LDS stores
    const float4 a0 = *reinterpret_cast<const float4*>(srcA + k0);
    const float4 a1 = *reinterpret_cast<const float4*>(srcA + k0 + 4);
    const float4 a2 = *reinterpret_cast<const float4*>(srcA + k0 + 8);
    const float4 a3 = *reinterpret_cast<const float4*>(srcA + k0 + 12);
    *reinterpret_cast<bf16x8*>(&As[sa_m][sa_k])     = cvt8(a0, a1);
    *reinterpret_cast<bf16x8*>(&As[sa_m][sa_k + 8]) = cvt8(a2, a3);
    // Stage B tile transposed layout == natural W row order (k-contiguous)
    const float4 b0 = *reinterpret_cast<const float4*>(srcB + k0);
    const float4 b1 = *reinterpret_cast<const float4*>(srcB + k0 + 4);
    *reinterpret_cast<bf16x8*>(&BsT[sb_n][sb_k]) = cvt8(b0, b1);
    if (k0 + 32 < EMBED) {
      __builtin_prefetch(srcA + k0 + 32, 0, 3);
      __builtin_prefetch(srcB + k0 + 32, 0, 3);
    }
    __syncthreads();

    const v16bf af = fragA(&As[w * 16][0], lane);
#pragma unroll
    for (int t = 0; t < 4; ++t)
      acc[t] = wmma_bf16(af, fragB(&BsT[0][0], t, lane), acc[t]);
    __syncthreads();
  }

  const int col   = lane & 15;
  const int rbase = (lane & 16) ? 8 : 0;
#pragma unroll
  for (int t = 0; t < 4; ++t) {
    const int gn = N0 + t * 16 + col;
    const float bsv = bias[gn];
#pragma unroll
    for (int r = 0; r < 8; ++r) {
      const int gm = M0 + w * 16 + rbase + r;
      Y[(size_t)gm * EMBED + gn] = (__bf16)((acc[t][r] + bsv) * scale);
    }
  }
}

// ---------------------------------------------------------------------------
// Kernel 2: raw masked scores = Qs @ K^T + mask (per batch), fp32 out.
// grid = (SEQ/64, SEQ/128, BATCH); block = 256
// ---------------------------------------------------------------------------
__global__ __launch_bounds__(256) void scores_kernel(
    const __bf16* __restrict__ Q, const __bf16* __restrict__ K,
    float* __restrict__ wts) {
  const int b = blockIdx.z;
  const __bf16* Qb = Q + (size_t)b * SEQ * EMBED;
  const __bf16* Kb = K + (size_t)b * SEQ * EMBED;
  float* wb = wts + (size_t)b * SEQ * SEQ;

  __shared__ __bf16 As[128][LDS_STRIDE];
  __shared__ __bf16 BsT[64][LDS_STRIDE];

  const int tid  = threadIdx.x;
  const int lane = tid & 31;
  const int w    = tid >> 5;
  const int M0 = blockIdx.y * 128;   // query rows
  const int N0 = blockIdx.x * 64;    // key cols

  const int sa_m = tid >> 1;
  const int sa_k = (tid & 1) * 16;
  const int sb_n = tid >> 2;
  const int sb_k = (tid & 3) * 8;

  v8f acc[4] = {};

  const __bf16* srcA = Qb + (size_t)(M0 + sa_m) * EMBED + sa_k;
  const __bf16* srcB = Kb + (size_t)(N0 + sb_n) * EMBED + sb_k;

  for (int k0 = 0; k0 < EMBED; k0 += 32) {
    *reinterpret_cast<bf16x8*>(&As[sa_m][sa_k]) =
        *reinterpret_cast<const bf16x8*>(srcA + k0);
    *reinterpret_cast<bf16x8*>(&As[sa_m][sa_k + 8]) =
        *reinterpret_cast<const bf16x8*>(srcA + k0 + 8);
    *reinterpret_cast<bf16x8*>(&BsT[sb_n][sb_k]) =
        *reinterpret_cast<const bf16x8*>(srcB + k0);
    if (k0 + 32 < EMBED) {
      __builtin_prefetch(srcA + k0 + 32, 0, 3);
      __builtin_prefetch(srcB + k0 + 32, 0, 3);
    }
    __syncthreads();

    const v16bf af = fragA(&As[w * 16][0], lane);
#pragma unroll
    for (int t = 0; t < 4; ++t)
      acc[t] = wmma_bf16(af, fragB(&BsT[0][0], t, lane), acc[t]);
    __syncthreads();
  }

  const int col   = lane & 15;
  const int rbase = (lane & 16) ? 8 : 0;
#pragma unroll
  for (int t = 0; t < 4; ++t) {
    const int gn = N0 + t * 16 + col;
#pragma unroll
    for (int r = 0; r < 8; ++r) {
      const int gm = M0 + w * 16 + rbase + r;   // query index
      wb[(size_t)gm * SEQ + gn] = acc[t][r] + ((gn > gm) ? -999.0f : 0.0f);
    }
  }
}

// ---------------------------------------------------------------------------
// Kernel 3: in-place row softmax over L=2048.  grid = B*L rows, block = 256.
// ---------------------------------------------------------------------------
__global__ __launch_bounds__(256) void softmax_kernel(float* __restrict__ wts) {
  float* p = wts + (size_t)blockIdx.x * SEQ;
  __shared__ float red[256];
  const int tid = threadIdx.x;

  float vals[8];
  float m = -3.0e38f;
#pragma unroll
  for (int j = 0; j < 8; ++j) {
    vals[j] = p[tid + j * 256];
    m = fmaxf(m, vals[j]);
  }
  red[tid] = m;
  __syncthreads();
  for (int s = 128; s > 0; s >>= 1) {
    if (tid < s) red[tid] = fmaxf(red[tid], red[tid + s]);
    __syncthreads();
  }
  m = red[0];
  __syncthreads();

  float sum = 0.0f;
#pragma unroll
  for (int j = 0; j < 8; ++j) {
    vals[j] = __expf(vals[j] - m);
    sum += vals[j];
  }
  red[tid] = sum;
  __syncthreads();
  for (int s = 128; s > 0; s >>= 1) {
    if (tid < s) red[tid] += red[tid + s];
    __syncthreads();
  }
  const float inv = 1.0f / red[0];
#pragma unroll
  for (int j = 0; j < 8; ++j) p[tid + j * 256] = vals[j] * inv;
}

// ---------------------------------------------------------------------------
// Kernel 4: out = wts @ V (per batch).  A = wts fp32 (->bf16), B = V bf16.
// V is [k][n] row-major so staging into BsT[n][k] transposes (scalar stores);
// fragment reads stay vectorized.
// grid = (EMBED/64, SEQ/128, BATCH); block = 256
// ---------------------------------------------------------------------------
__global__ __launch_bounds__(256) void out_kernel(
    const float* __restrict__ wts, const __bf16* __restrict__ V,
    float* __restrict__ out) {
  const int b = blockIdx.z;
  const float*  Ab = wts + (size_t)b * SEQ * SEQ;
  const __bf16* Vb = V + (size_t)b * SEQ * EMBED;
  float* ob = out + (size_t)b * SEQ * EMBED;

  __shared__ __bf16 As[128][LDS_STRIDE];
  __shared__ __bf16 BsT[64][LDS_STRIDE];

  const int tid  = threadIdx.x;
  const int lane = tid & 31;
  const int w    = tid >> 5;
  const int M0 = blockIdx.y * 128;   // query rows
  const int N0 = blockIdx.x * 64;    // embed cols

  const int sa_m = tid >> 1;
  const int sa_k = (tid & 1) * 16;
  const int sb_k = tid >> 3;           // 0..31 (V row within K tile)
  const int sb_n = (tid & 7) * 8;      // 0..56

  v8f acc[4] = {};

  const float*  srcA = Ab + (size_t)(M0 + sa_m) * SEQ + sa_k;
  const __bf16* srcB = Vb + (size_t)sb_k * EMBED + N0 + sb_n;

  for (int k0 = 0; k0 < SEQ; k0 += 32) {
    const float4 a0 = *reinterpret_cast<const float4*>(srcA + k0);
    const float4 a1 = *reinterpret_cast<const float4*>(srcA + k0 + 4);
    const float4 a2 = *reinterpret_cast<const float4*>(srcA + k0 + 8);
    const float4 a3 = *reinterpret_cast<const float4*>(srcA + k0 + 12);
    *reinterpret_cast<bf16x8*>(&As[sa_m][sa_k])     = cvt8(a0, a1);
    *reinterpret_cast<bf16x8*>(&As[sa_m][sa_k + 8]) = cvt8(a2, a3);
    const bf16x8 vv =
        *reinterpret_cast<const bf16x8*>(srcB + (size_t)k0 * EMBED);
#pragma unroll
    for (int j = 0; j < 8; ++j) BsT[sb_n + j][sb_k] = vv[j];
    if (k0 + 32 < SEQ) {
      __builtin_prefetch(srcA + k0 + 32, 0, 3);
      __builtin_prefetch(srcB + (size_t)(k0 + 32) * EMBED, 0, 3);
    }
    __syncthreads();

    const v16bf af = fragA(&As[w * 16][0], lane);
#pragma unroll
    for (int t = 0; t < 4; ++t)
      acc[t] = wmma_bf16(af, fragB(&BsT[0][0], t, lane), acc[t]);
    __syncthreads();
  }

  const int col   = lane & 15;
  const int rbase = (lane & 16) ? 8 : 0;
#pragma unroll
  for (int t = 0; t < 4; ++t) {
    const int gn = N0 + t * 16 + col;
#pragma unroll
    for (int r = 0; r < 8; ++r) {
      const int gm = M0 + w * 16 + rbase + r;
      ob[(size_t)gm * EMBED + gn] = acc[t][r];
    }
  }
}

// ---------------------------------------------------------------------------
extern "C" void kernel_launch(void* const* d_in, const int* in_sizes, int n_in,
                              void* d_out, int out_size, void* d_ws, size_t ws_size,
                              hipStream_t stream) {
  (void)in_sizes; (void)n_in; (void)out_size; (void)ws_size;

  const float* x  = (const float*)d_in[0];
  const float* Wq = (const float*)d_in[1];
  const float* bq = (const float*)d_in[2];
  const float* Wk = (const float*)d_in[3];
  const float* bk = (const float*)d_in[4];
  const float* Wv = (const float*)d_in[5];
  const float* bv = (const float*)d_in[6];

  float* out = (float*)d_out;                               // (B,L,d) fp32
  float* wts = out + (size_t)BATCH * SEQ * EMBED;           // (B,L,L) fp32

  __bf16* qkv = (__bf16*)d_ws;                              // 3 * 16 MB bf16
  const size_t planeSz = (size_t)MROWS * EMBED;
  const __bf16* Q = qkv;
  const __bf16* K = qkv + planeSz;
  const __bf16* V = qkv + 2 * planeSz;

  dim3 blk(256);

  // 1) QKV projection (Q pre-scaled by 1/sqrt(d))
  qkv_proj_kernel<<<dim3(EMBED / 64, MROWS / 128, 3), blk, 0, stream>>>(
      x, Wq, bq, Wk, bk, Wv, bv, qkv);

  // 2) masked raw scores into wts region
  scores_kernel<<<dim3(SEQ / 64, SEQ / 128, BATCH), blk, 0, stream>>>(Q, K, wts);

  // 3) row softmax in place
  softmax_kernel<<<dim3(BATCH * SEQ), blk, 0, stream>>>(wts);

  // 4) out = wts @ V
  out_kernel<<<dim3(EMBED / 64, SEQ / 128, BATCH), blk, 0, stream>>>(wts, V, out);
}
